// ThreeLayerCNN_83004537963207
// MI455X (gfx1250) — matrix operand order repaired and checked by reading, not compiled, not run on hardware
//
#include <hip/hip_runtime.h>
#include <hip/hip_bf16.h>
#include <math.h>

// ---------------------------------------------------------------------------
// Types for WMMA (CDNA5 / gfx1250, wave32)
// ---------------------------------------------------------------------------
typedef __attribute__((ext_vector_type(16))) _Float16 v16h;
typedef __attribute__((ext_vector_type(8)))  float    v8f;
typedef __attribute__((ext_vector_type(4)))  float    v4f;

union V16H { v4f f[2]; v16h h; };

// Load a 16-half WMMA fragment as two 16-byte chunks (works for global & LDS).
__device__ __forceinline__ v16h ld_frag16(const _Float16* p0, const _Float16* p1) {
    V16H u;
    u.f[0] = *reinterpret_cast<const v4f*>(p0);
    u.f[1] = *reinterpret_cast<const v4f*>(p1);
    return u.h;
}

// ---------------------------------------------------------------------------
// Problem constants
// ---------------------------------------------------------------------------
#define BATCH 512
#define RROWS 692                   // 500 (l11) + 166 (l12) + 1 (c32) + 25 (c3d)
#define NROWS (BATCH * RROWS)       // 354304 head rows, divisible by 16
#define KPAD  64                    // head K = 50 padded to 64
#define N1PAD 512                   // head N1 = 500 padded to 512
#define N2PAD 16                    // head N2 = 10 padded to 16
#define CKPAD 512                   // conv K = 450 padded to 512
#define CNPAD 64                    // conv N = 50 padded to 64

// ---------------------------------------------------------------------------
// Scalar support kernels
// ---------------------------------------------------------------------------

// conv1 (1->50, 3x3, valid) + relu + maxpool2 : x[512,1,28,28] -> h1[512,50,13,13]
__global__ void k_conv1_relu_pool(const float* __restrict__ x,
                                  const float* __restrict__ w,
                                  const float* __restrict__ bias,
                                  float* __restrict__ h1) {
    int idx = blockIdx.x * blockDim.x + threadIdx.x;
    if (idx >= BATCH * 50 * 13 * 13) return;
    int px = idx % 13; int t = idx / 13;
    int py = t % 13;   t /= 13;
    int c  = t % 50;   int b = t / 50;
    const float* xb = x + (size_t)b * 784;
    const float* wc = w + c * 9;
    float bi = bias[c];
    float m = -__builtin_inff();
    #pragma unroll
    for (int dy = 0; dy < 2; ++dy)
    #pragma unroll
    for (int dx = 0; dx < 2; ++dx) {
        int oy = 2 * py + dy, ox = 2 * px + dx;
        float a = bi;
        #pragma unroll
        for (int ky = 0; ky < 3; ++ky)
        #pragma unroll
        for (int kx = 0; kx < 3; ++kx)
            a = fmaf(xb[(oy + ky) * 28 + ox + kx], wc[ky * 3 + kx], a);
        a = a > 0.f ? a : 0.f;
        m = fmaxf(m, a);
    }
    h1[idx] = m;
}

// router: d[b] = argmax_j ( feat[b,:] . w[j,:] + bias[j] ), first max on ties
__global__ void k_router(const float* __restrict__ feat,
                         const float* __restrict__ w,
                         const float* __restrict__ bias,
                         int* __restrict__ dout, int F, int J) {
    __shared__ float red[3 * 256];
    int b = blockIdx.x, t = threadIdx.x;
    float s[3] = {0.f, 0.f, 0.f};
    const float* fb = feat + (size_t)b * F;
    for (int i = t; i < F; i += 256) {
        float v = fb[i];
        for (int j = 0; j < J; ++j) s[j] = fmaf(v, w[(size_t)j * F + i], s[j]);
    }
    for (int j = 0; j < J; ++j) red[j * 256 + t] = s[j];
    __syncthreads();
    for (int off = 128; off > 0; off >>= 1) {
        if (t < off)
            for (int j = 0; j < J; ++j) red[j * 256 + t] += red[j * 256 + t + off];
        __syncthreads();
    }
    if (t == 0) {
        float best = -__builtin_inff(); int bi = 0;
        for (int j = 0; j < J; ++j) {
            float v = red[j * 256] + bias[j];
            if (v > best) { best = v; bi = j; }
        }
        dout[b] = bi;
    }
}

// maxpool 7x7 on h1 -> p7[512,50]
__global__ void k_pool7(const float* __restrict__ h1, float* __restrict__ p7) {
    int idx = blockIdx.x * blockDim.x + threadIdx.x;
    if (idx >= BATCH * 50) return;
    const float* src = h1 + (size_t)idx * 169;
    float m = -__builtin_inff();
    for (int y = 0; y < 7; ++y)
        for (int x = 0; x < 7; ++x)
            m = fmaxf(m, src[y * 13 + x]);
    p7[idx] = m;
}

// c33[b,c,i,j] = sum_k h1[b,c,i,k] * c1c3_w[j,k] + c1c3_b[j]
__global__ void k_lin13(const float* __restrict__ h1,
                        const float* __restrict__ w,
                        const float* __restrict__ bias,
                        float* __restrict__ c33) {
    int idx = blockIdx.x * blockDim.x + threadIdx.x;
    if (idx >= BATCH * 50 * 13 * 13) return;
    int j = idx % 13;
    const float* row = h1 + (size_t)(idx / 13) * 13;
    float a = bias[j];
    #pragma unroll
    for (int k = 0; k < 13; ++k) a = fmaf(row[k], w[j * 13 + k], a);
    c33[idx] = a;
}

// ---------------------------------------------------------------------------
// Implicit-GEMM conv path (conv2 / conv3): im2col -> WMMA GEMM -> pool
// ---------------------------------------------------------------------------

// im2col: in[512,50,H,H] -> A[M=512*OH*OH][512] f16, col k = ic*9 + ky*3 + kx
__global__ void k_im2col(const float* __restrict__ in, _Float16* __restrict__ A,
                         int H, int OH) {
    long idx = (long)blockIdx.x * blockDim.x + threadIdx.x;
    long total = (long)BATCH * OH * OH * CKPAD;
    if (idx >= total) return;
    int k = (int)(idx & (CKPAD - 1));
    long m = idx >> 9;
    float v = 0.f;
    if (k < 450) {
        int ic = k / 9, r = k % 9;
        int ky = r / 3, kx = r % 3;
        int ox = (int)(m % OH); long t = m / OH;
        int oy = (int)(t % OH); int b = (int)(t / OH);
        v = in[((size_t)(b * 50 + ic) * H + oy + ky) * H + ox + kx];
    }
    A[idx] = (_Float16)v;
}

// pad+convert conv weight (50,50,3,3) -> [64][512] f16 (row oc, col ic*9+ky*3+kx)
__global__ void k_cvtWc(const float* __restrict__ w, _Float16* __restrict__ Wh) {
    int idx = blockIdx.x * blockDim.x + threadIdx.x;
    if (idx >= CNPAD * CKPAD) return;
    int k = idx & (CKPAD - 1), oc = idx >> 9;
    Wh[idx] = (_Float16)((oc < 50 && k < 450) ? w[(size_t)oc * 450 + k] : 0.f);
}

// C[M,64] = relu(A[M,512] @ Wh[64,512]^T + bias). One wave = 16x64 output tile.
__global__ __launch_bounds__(64) void k_conv_wmma(const _Float16* __restrict__ A,
                                                  const _Float16* __restrict__ Wh,
                                                  const float* __restrict__ bias,
                                                  float* __restrict__ C, int M) {
    const int wv   = threadIdx.x >> 5;
    const int lane = threadIdx.x & 31;
    const int half = lane >> 4;
    const int ln   = lane & 15;
    const int m0   = (blockIdx.x * 2 + wv) * 16;
    (void)M;

    v8f acc0 = {}, acc1 = {}, acc2 = {}, acc3 = {};
    const _Float16* arow = A + (size_t)(m0 + ln) * CKPAD + half * 16;
    const _Float16* brow = Wh + (size_t)ln * CKPAD + half * 16;

    #pragma unroll
    for (int ks = 0; ks < 16; ++ks) {
        // A frag: 16 contiguous halves at kb + half*16
        v16h aA = ld_frag16(arow + ks * 32, arow + ks * 32 + 8);
        // B frags: row n = nt*16+ln, 16 contiguous halves at kb + half*16
        v16h b0 = ld_frag16(brow + ks * 32,              brow + ks * 32 + 8);
        v16h b1 = ld_frag16(brow + 16 * CKPAD + ks * 32, brow + 16 * CKPAD + ks * 32 + 8);
        v16h b2 = ld_frag16(brow + 32 * CKPAD + ks * 32, brow + 32 * CKPAD + ks * 32 + 8);
        v16h b3 = ld_frag16(brow + 48 * CKPAD + ks * 32, brow + 48 * CKPAD + ks * 32 + 8);
        acc0 = __builtin_amdgcn_wmma_f32_16x16x32_f16(false, aA, false, b0, (short)0, acc0, false, false);
        acc1 = __builtin_amdgcn_wmma_f32_16x16x32_f16(false, aA, false, b1, (short)0, acc1, false, false);
        acc2 = __builtin_amdgcn_wmma_f32_16x16x32_f16(false, aA, false, b2, (short)0, acc2, false, false);
        acc3 = __builtin_amdgcn_wmma_f32_16x16x32_f16(false, aA, false, b3, (short)0, acc3, false, false);
    }

    // Epilogue: C layout row = v + 8*half, col = nt*16 + ln. relu(x + bias).
    v8f* accs[4] = {&acc0, &acc1, &acc2, &acc3};
    #pragma unroll
    for (int nt = 0; nt < 4; ++nt) {
        int n = nt * 16 + ln;
        float bi = (n < 50) ? bias[n] : 0.f;
        #pragma unroll
        for (int v = 0; v < 8; ++v) {
            float r = (*accs[nt])[v] + bi;
            r = r > 0.f ? r : 0.f;
            C[(size_t)(m0 + v + half * 8) * CNPAD + n] = r;
        }
    }
}

// pool 2x2 stride 2 over GEMM output C[(b*OH+oy)*OH+ox][64] -> out[512,50,PH,PH]
__global__ void k_pool_from_gemm(const float* __restrict__ C, float* __restrict__ out,
                                 int OH, int PH) {
    int idx = blockIdx.x * blockDim.x + threadIdx.x;
    int total = BATCH * 50 * PH * PH;
    if (idx >= total) return;
    int px = idx % PH; int t = idx / PH;
    int py = t % PH;   t /= PH;
    int oc = t % 50;   int b = t / 50;
    float m = -__builtin_inff();
    #pragma unroll
    for (int dy = 0; dy < 2; ++dy)
    #pragma unroll
    for (int dx = 0; dx < 2; ++dx) {
        size_t row = (size_t)b * OH * OH + (size_t)(2 * py + dy) * OH + (2 * px + dx);
        m = fmaxf(m, C[row * CNPAD + oc]);
    }
    out[idx] = m;
}

// ---------------------------------------------------------------------------
// Branch fill kernels (build packed head input Hin[B*692][64] f16)
// ---------------------------------------------------------------------------

// Hin rows [0,500): l11 branch. row r: c=r/10, j0=(r%10)*50
__global__ void k_fill_l11(const float* __restrict__ p7,
                           const float* __restrict__ w,   // c1l1_w (500,1)
                           const float* __restrict__ bias,
                           _Float16* __restrict__ Hin) {
    int idx = blockIdx.x * blockDim.x + threadIdx.x;
    if (idx >= BATCH * 500) return;
    int r = idx % 500, b = idx / 500;
    int c = r / 10, j0 = (r % 10) * 50;
    float pv = p7[b * 50 + c];
    _Float16* dst = Hin + ((size_t)b * RROWS + r) * KPAD;
    for (int k = 0; k < KPAD; ++k) {
        float v = (k < 50) ? fmaf(pv, w[j0 + k], bias[j0 + k]) : 0.f;
        dst[k] = (_Float16)v;
    }
}

// Hin rows [500,666): l12 branch = maxpool3 of Linear(5->500) on h2
__global__ void k_fill_l12(const float* __restrict__ h2,
                           const float* __restrict__ w,   // c2l1_w (500,5)
                           const float* __restrict__ bias,
                           _Float16* __restrict__ Hin) {
    int idx = blockIdx.x * blockDim.x + threadIdx.x;
    if (idx >= BATCH * 166) return;
    int r = idx % 166, b = idx / 166;
    _Float16* dst = Hin + ((size_t)b * RROWS + 500 + r) * KPAD;
    for (int k = 0; k < KPAD; ++k) {
        float v = 0.f;
        if (k < 50) {
            int flat = r * 50 + k;
            int c = flat / 166, t = flat % 166;
            float mv = -__builtin_inff();
            for (int i = 0; i < 3; ++i) {
                const float* hr = h2 + ((size_t)(b * 50 + c) * 5 + i) * 5;
                for (int dj = 0; dj < 3; ++dj) {
                    int j = 3 * t + dj;
                    float s = bias[j];
                    #pragma unroll
                    for (int m = 0; m < 5; ++m) s = fmaf(hr[m], w[j * 5 + m], s);
                    mv = fmaxf(mv, s);
                }
            }
            v = mv;
        }
        dst[k] = (_Float16)v;
    }
}

// Hin row 666: c32 branch (channels as columns)
__global__ void k_fill_c32(const float* __restrict__ c32, _Float16* __restrict__ Hin) {
    int idx = blockIdx.x * blockDim.x + threadIdx.x;
    if (idx >= BATCH * KPAD) return;
    int k = idx % KPAD, b = idx / KPAD;
    Hin[((size_t)b * RROWS + 666) * KPAD + k] =
        (_Float16)((k < 50) ? c32[b * 50 + k] : 0.f);
}

// Hin rows [667,692): c3d branch [B,50,5,5] -> [B,25,50] row-major flatten
__global__ void k_fill_c3d(const float* __restrict__ c3d, _Float16* __restrict__ Hin) {
    int idx = blockIdx.x * blockDim.x + threadIdx.x;
    if (idx >= BATCH * 25 * KPAD) return;
    int k = idx % KPAD; int t = idx / KPAD;
    int r = t % 25, b = t / 25;
    float v = 0.f;
    if (k < 50) {
        int flat = r * 50 + k;
        int c = flat / 25, pos = flat % 25;
        v = c3d[((size_t)(b * 50 + c) * 5 + pos / 5) * 5 + pos % 5];
    }
    Hin[((size_t)b * RROWS + 667 + r) * KPAD + k] = (_Float16)v;
}

// pad+convert l1_w (500,50) -> W1h [512][64] f16, l2_w (10,500) -> W2h [16][512]
__global__ void k_cvtW1(const float* __restrict__ w, _Float16* __restrict__ W1h) {
    int idx = blockIdx.x * blockDim.x + threadIdx.x;
    if (idx >= N1PAD * KPAD) return;
    int k = idx % KPAD, n = idx / KPAD;
    W1h[idx] = (_Float16)((n < 500 && k < 50) ? w[n * 50 + k] : 0.f);
}
__global__ void k_cvtW2(const float* __restrict__ w, _Float16* __restrict__ W2h) {
    int idx = blockIdx.x * blockDim.x + threadIdx.x;
    if (idx >= N2PAD * N1PAD) return;
    int k = idx % N1PAD, n = idx / N1PAD;
    W2h[idx] = (_Float16)((n < 10 && k < 500) ? w[n * 500 + k] : 0.f);
}

// copy routing decisions into d_out tail (stored as int32 bit patterns)
__global__ void k_write_routes(const int* __restrict__ d1, const int* __restrict__ d2,
                               float* __restrict__ out) {
    int i = blockIdx.x * blockDim.x + threadIdx.x;
    if (i >= BATCH) return;
    int* o = reinterpret_cast<int*>(out) + (size_t)NROWS * 10;
    o[i] = d1[i];
    o[BATCH + i] = d2[i];
}

// ---------------------------------------------------------------------------
// Fused head kernel (21 GFLOP hot path):
//   t1 = relu(Hin @ W1^T + b1) via v_wmma_f32_16x16x32_f16, staged f16 in LDS
//   lg = t1 @ W2^T + b2        via v_wmma_f32_16x16x32_f16
//   out = log_softmax(lg[:, :10]) * routing_mask
// ---------------------------------------------------------------------------
__global__ __launch_bounds__(64) void k_head_wmma(
    const _Float16* __restrict__ Hin, const _Float16* __restrict__ W1h,
    const _Float16* __restrict__ W2h, const float* __restrict__ b1,
    const float* __restrict__ b2, const int* __restrict__ d1,
    const int* __restrict__ d2, float* __restrict__ out) {

    __shared__ _Float16 t1[2][16][N1PAD];   // 32 KB
    __shared__ float    lg[2][16][16];      // 2 KB

    const int wv   = threadIdx.x >> 5;
    const int lane = threadIdx.x & 31;
    const int half = lane >> 4;
    const int ln   = lane & 15;
    const int m0   = (blockIdx.x * 2 + wv) * 16;

    // A fragments for GEMM1 (K = 64 -> two 16x16x32 steps); reused for all n-tiles.
    const _Float16* arow = Hin + (size_t)(m0 + ln) * KPAD + half * 16;
    v16h a0 = ld_frag16(arow,      arow + 8);
    v16h a1 = ld_frag16(arow + 32, arow + 40);

    // GEMM1 over 32 n-tiles of W1 (N=512)
    for (int nt = 0; nt < 32; ++nt) {
        int nG = nt * 16 + ln;
        const _Float16* brow = W1h + (size_t)nG * KPAD + half * 16;
        v16h b0k = ld_frag16(brow,      brow + 8);
        v16h b1k = ld_frag16(brow + 32, brow + 40);
        v8f acc = {};
        acc = __builtin_amdgcn_wmma_f32_16x16x32_f16(false, a0, false, b0k,
                                                     (short)0, acc, false, false);
        acc = __builtin_amdgcn_wmma_f32_16x16x32_f16(false, a1, false, b1k,
                                                     (short)0, acc, false, false);
        float bias = (nG < 500) ? b1[nG] : 0.f;
        #pragma unroll
        for (int v = 0; v < 8; ++v) {        // C layout: row v + 8*half, col ln
            float r = acc[v] + bias;
            r = r > 0.f ? r : 0.f;
            t1[wv][v + half * 8][nt * 16 + ln] = (_Float16)r;
        }
    }
    __syncthreads();

    // GEMM2: [16,512] x [512,16] over 16 k-steps
    v8f acc2 = {};
    #pragma unroll
    for (int ks = 0; ks < 16; ++ks) {
        const _Float16* ap = &t1[wv][ln][ks * 32 + half * 16];
        v16h aA = ld_frag16(ap, ap + 8);
        const _Float16* bp = W2h + (size_t)ln * N1PAD + ks * 32 + half * 16;
        v16h bB = ld_frag16(bp, bp + 8);
        acc2 = __builtin_amdgcn_wmma_f32_16x16x32_f16(false, aA, false, bB,
                                                      (short)0, acc2, false, false);
    }
    #pragma unroll
    for (int v = 0; v < 8; ++v)
        lg[wv][v + half * 8][ln] = acc2[v] + ((ln < 10) ? b2[ln] : 0.f);
    __syncthreads();

    // Epilogue: per-row log_softmax over 10 classes + routing mask
    if (lane < 16) {
        int row = m0 + lane;
        int b = row / RROWS;
        int r = row - b * RROWS;
        int dd1 = d1[b], dd2 = d2[b];
        float mask;
        if (r < 500)       mask = (dd1 == 2) ? 1.f : 0.f;
        else if (r < 666)  mask = (dd1 == 0 && dd2 == 1) ? 1.f : 0.f;
        else if (r == 666) mask = (dd1 == 0 && dd2 == 0) ? 1.f : 0.f;
        else               mask = (dd1 == 1) ? 1.f : 0.f;
        float mx = -__builtin_inff();
        for (int j = 0; j < 10; ++j) mx = fmaxf(mx, lg[wv][lane][j]);
        float se = 0.f;
        for (int j = 0; j < 10; ++j) se += __expf(lg[wv][lane][j] - mx);
        float lse = mx + __logf(se);
        float* po = out + (size_t)row * 10;
        for (int j = 0; j < 10; ++j) po[j] = (lg[wv][lane][j] - lse) * mask;
    }
}

// ---------------------------------------------------------------------------
// Host-side launch
// ---------------------------------------------------------------------------
extern "C" void kernel_launch(void* const* d_in, const int* in_sizes, int n_in,
                              void* d_out, int out_size, void* d_ws, size_t ws_size,
                              hipStream_t stream) {
    (void)in_sizes; (void)n_in; (void)out_size; (void)ws_size;

    const float* x      = (const float*)d_in[0];
    const float* c1_w   = (const float*)d_in[1];
    const float* c1_b   = (const float*)d_in[2];
    const float* c2_w   = (const float*)d_in[3];
    const float* c2_b   = (const float*)d_in[4];
    const float* c3_w   = (const float*)d_in[5];
    const float* c3_b   = (const float*)d_in[6];
    const float* l1_w   = (const float*)d_in[7];
    const float* l1_b   = (const float*)d_in[8];
    const float* l2_w   = (const float*)d_in[9];
    const float* l2_b   = (const float*)d_in[10];
    const float* d1_w   = (const float*)d_in[11];
    const float* d1_b   = (const float*)d_in[12];
    const float* d2_w   = (const float*)d_in[13];
    const float* d2_b   = (const float*)d_in[14];
    const float* c1c3_w = (const float*)d_in[15];
    const float* c1c3_b = (const float*)d_in[16];
    const float* c1l1_w = (const float*)d_in[17];
    const float* c1l1_b = (const float*)d_in[18];
    const float* c2l1_w = (const float*)d_in[19];
    const float* c2l1_b = (const float*)d_in[20];

    float* out = (float*)d_out;

    // workspace carve-up (256B aligned)
    char* base = (char*)d_ws;
    size_t off = 0;
    auto carve = [&](size_t bytes) -> char* {
        char* p = base + off;
        off = (off + bytes + 255) & ~(size_t)255;
        return p;
    };
    const long M2 = (long)BATCH * 11 * 11;   // 61952 pixels for 13->11 convs
    const long M3 = (long)BATCH * 3 * 3;     // 4608 pixels for 5->3 conv

    float*     h1   = (float*)carve((size_t)BATCH * 50 * 169 * 4);   // 17.3 MB
    float*     c33  = (float*)carve((size_t)BATCH * 50 * 169 * 4);   // 17.3 MB
    float*     h2   = (float*)carve((size_t)BATCH * 50 * 25 * 4);    // 2.56 MB
    float*     c3d  = (float*)carve((size_t)BATCH * 50 * 25 * 4);    // 2.56 MB
    float*     c32  = (float*)carve((size_t)BATCH * 50 * 4);
    float*     p7   = (float*)carve((size_t)BATCH * 50 * 4);
    int*       d1b  = (int*)carve(BATCH * 4);
    int*       d2b  = (int*)carve(BATCH * 4);
    _Float16*  Hin  = (_Float16*)carve((size_t)NROWS * KPAD * 2);    // 45.4 MB
    _Float16*  W1h  = (_Float16*)carve((size_t)N1PAD * KPAD * 2);
    _Float16*  W2h  = (_Float16*)carve((size_t)N2PAD * N1PAD * 2);
    _Float16*  Wc2h = (_Float16*)carve((size_t)CNPAD * CKPAD * 2);
    _Float16*  Wc3h = (_Float16*)carve((size_t)CNPAD * CKPAD * 2);
    _Float16*  Acol = (_Float16*)carve((size_t)M2 * CKPAD * 2);      // 63.4 MB (reused)
    float*     Cg   = (float*)carve((size_t)M2 * CNPAD * 4);         // 15.9 MB (reused)

    const int T = 256;
    auto G = [](long n, int t) { return (int)((n + t - 1) / t); };

    // weight conversions (independent)
    k_cvtW1<<<G(N1PAD * KPAD, T), T, 0, stream>>>(l1_w, W1h);
    k_cvtW2<<<G(N2PAD * N1PAD, T), T, 0, stream>>>(l2_w, W2h);
    k_cvtWc<<<G(CNPAD * CKPAD, T), T, 0, stream>>>(c2_w, Wc2h);
    k_cvtWc<<<G(CNPAD * CKPAD, T), T, 0, stream>>>(c3_w, Wc3h);

    // stage 1
    k_conv1_relu_pool<<<G((long)BATCH * 50 * 169, T), T, 0, stream>>>(x, c1_w, c1_b, h1);
    k_router<<<BATCH, T, 0, stream>>>(h1, d1_w, d1_b, d1b, 8450, 3);
    k_pool7<<<G(BATCH * 50, T), T, 0, stream>>>(h1, p7);
    k_lin13<<<G((long)BATCH * 50 * 169, T), T, 0, stream>>>(h1, c1c3_w, c1c3_b, c33);
    k_fill_l11<<<G((long)BATCH * 500, T), T, 0, stream>>>(p7, c1l1_w, c1l1_b, Hin);

    // conv2: h1 -> h2 (implicit GEMM on WMMA)
    k_im2col<<<G(M2 * CKPAD, T), T, 0, stream>>>(h1, Acol, 13, 11);
    k_conv_wmma<<<(int)(M2 / 32), 64, 0, stream>>>(Acol, Wc2h, c2_b, Cg, (int)M2);
    k_pool_from_gemm<<<G((long)BATCH * 50 * 25, T), T, 0, stream>>>(Cg, h2, 11, 5);
    k_router<<<BATCH, T, 0, stream>>>(h2, d2_w, d2_b, d2b, 1250, 2);

    // conv3 on c33 -> c3d
    k_im2col<<<G(M2 * CKPAD, T), T, 0, stream>>>(c33, Acol, 13, 11);
    k_conv_wmma<<<(int)(M2 / 32), 64, 0, stream>>>(Acol, Wc3h, c3_b, Cg, (int)M2);
    k_pool_from_gemm<<<G((long)BATCH * 50 * 25, T), T, 0, stream>>>(Cg, c3d, 11, 5);

    // conv3 on h2 -> c32
    k_im2col<<<G(M3 * CKPAD, T), T, 0, stream>>>(h2, Acol, 5, 3);
    k_conv_wmma<<<(int)(M3 / 32), 64, 0, stream>>>(Acol, Wc3h, c3_b, Cg, (int)M3);
    k_pool_from_gemm<<<G((long)BATCH * 50 * 1, T), T, 0, stream>>>(Cg, c32, 3, 1);

    // remaining branch fills
    k_fill_l12<<<G((long)BATCH * 166, T), T, 0, stream>>>(h2, c2l1_w, c2l1_b, Hin);
    k_fill_c32<<<G((long)BATCH * KPAD, T), T, 0, stream>>>(c32, Hin);
    k_fill_c3d<<<G((long)BATCH * 25 * KPAD, T), T, 0, stream>>>(c3d, Hin);

    // fused WMMA head over all 354304 rows (32 rows per 64-thread block)
    k_head_wmma<<<NROWS / 32, 64, 0, stream>>>(Hin, W1h, W2h, l1_b, l2_b, d1b, d2b, out);

    // routing decisions appended to output
    k_write_routes<<<G(BATCH, T), T, 0, stream>>>(d1b, d2b, out);
}
